// ClassEmbedding_11373073399914
// MI455X (gfx1250) — matrix operand order repaired and checked by reading, not compiled, not run on hardware
//
#include <hip/hip_runtime.h>
#include <hip/hip_bf16.h>
#include <stdint.h>

// ---------------------------------------------------------------------------
// Sizes (fixed by the reference)
// ---------------------------------------------------------------------------
#define BN      12800          // 64 * 200
#define TSTEPS  8
#define EMB     300
#define EMBP    320            // padded to multiple of 32 (WMMA K step)
#define RNN     512
#define NG      2048           // 4 * RNN
#define K1      (EMBP + RNN)   // 832  (layer-1 concat K)
#define K2      (RNN + RNN)    // 1024 (layer-2 concat K)

typedef __attribute__((ext_vector_type(16))) __bf16 v16bf;
typedef __attribute__((ext_vector_type(8)))  float  v8f;
typedef int v4i __attribute__((vector_size(16)));   // matches builtin param type

// CDNA5 async global->LDS copies (ASYNCcnt-tracked), with guarded fallback.
#if __has_builtin(__builtin_amdgcn_global_load_async_to_lds_b128)
#define USE_ASYNC_LDS 1
#else
#define USE_ASYNC_LDS 0
#endif

#if USE_ASYNC_LDS
#if __has_builtin(__builtin_amdgcn_s_wait_asynccnt)
#define WAIT_ASYNCCNT(n) __builtin_amdgcn_s_wait_asynccnt(n)
#else
#define WAIT_ASYNCCNT(n) asm volatile("s_wait_asynccnt %0" ::"n"(n) : "memory")
#endif
// Builtin signature (from probe diagnostics): (v4i AS1* gsrc, v4i AS3* ldsdst,
// imm offset, imm cpol)
#define GLD_ASYNC_B128(ldsdst, gsrc)                                          \
    __builtin_amdgcn_global_load_async_to_lds_b128(                           \
        (__attribute__((address_space(1))) v4i*)(gsrc),                       \
        (__attribute__((address_space(3))) v4i*)(ldsdst), 0, 0)
#endif

static __device__ __forceinline__ unsigned short f2bf(float f) {
    unsigned int u = __float_as_uint(f);
    unsigned int r = (u + 0x7FFFu + ((u >> 16) & 1u)) >> 16;   // RNE bf16
    return (unsigned short)r;
}
static __device__ __forceinline__ float sigmoidf(float x) {
    return 1.0f / (1.0f + __expf(-x));
}

// ---------------------------------------------------------------------------
// Weight packing: Wc1[2048, 832] = [W_ih1 (300, zero-pad to 320) | W_hh1 (512)]
// ---------------------------------------------------------------------------
__global__ void pack_w1_kernel(const float* __restrict__ Wih,
                               const float* __restrict__ Whh,
                               unsigned short* __restrict__ Wc, int total) {
    int idx = blockIdx.x * blockDim.x + threadIdx.x;
    if (idx >= total) return;
    int row = idx / K1, col = idx - row * K1;
    float v = 0.0f;
    if (col < EMB)        v = Wih[row * EMB + col];
    else if (col >= EMBP) v = Whh[row * RNN + (col - EMBP)];
    Wc[idx] = f2bf(v);
}

// Wc2[2048, 1024] = [W_ih2 (512) | W_hh2 (512)]
__global__ void pack_w2_kernel(const float* __restrict__ Wih,
                               const float* __restrict__ Whh,
                               unsigned short* __restrict__ Wc, int total) {
    int idx = blockIdx.x * blockDim.x + threadIdx.x;
    if (idx >= total) return;
    int row = idx / K2, col = idx - row * K2;
    float v = (col < RNN) ? Wih[row * RNN + col] : Whh[row * RNN + (col - RNN)];
    Wc[idx] = f2bf(v);
}

__global__ void bias_combine_kernel(const float* __restrict__ bih1, const float* __restrict__ bhh1,
                                    const float* __restrict__ bih2, const float* __restrict__ bhh2,
                                    float* __restrict__ b1, float* __restrict__ b2) {
    int idx = blockIdx.x * blockDim.x + threadIdx.x;
    if (idx >= NG) return;
    b1[idx] = bih1[idx] + bhh1[idx];
    b2[idx] = bih2[idx] + bhh2[idx];
}

__global__ void zero16_kernel(uint4* __restrict__ p, int total) {
    int idx = blockIdx.x * blockDim.x + threadIdx.x;
    if (idx >= total) return;
    p[idx] = make_uint4(0u, 0u, 0u, 0u);
}

// ---------------------------------------------------------------------------
// Embedding gather + tanh -> bf16 into Z1[:, 0:300] for timestep t
// ---------------------------------------------------------------------------
__global__ void embed_tanh_kernel(const int* __restrict__ sent,
                                  const float* __restrict__ w2v,
                                  unsigned short* __restrict__ Z1,
                                  int t, int total) {
    int idx = blockIdx.x * blockDim.x + threadIdx.x;
    if (idx >= total) return;
    int bn = idx / EMB, e = idx - bn * EMB;
    int id = sent[bn * TSTEPS + t];
    float x = tanhf(w2v[(size_t)id * EMB + e]);
    Z1[(size_t)bn * K1 + e] = f2bf(x);
}

// ---------------------------------------------------------------------------
// GEMM: gates[M, NG] = A[M, K](bf16) * Bw[NG, K](bf16)^T + bias[NG]
// 256 threads = 8 waves (4M x 2N), block tile 128(M) x 64(N), wave tile 32x32.
// K is consumed in chunks of 64: A/B tiles are staged in double-buffered LDS
// via async global->LDS b128 copies; fragments read back with ds_load_b128 in
// the exact CDNA5 WMMA per-lane layout; 8x v_wmma_f32_16x16x32_bf16 per stage.
// Requires M%128==0, N==2048, K%64==0. No divergence -> EXEC all-ones.
// ---------------------------------------------------------------------------
#define KT    64              // K elements per LDS stage
#define ASTR  72              // padded LDS row stride in elements (144B = 9*16B)

__global__ void __launch_bounds__(256)
gemm_bf16_wmma_kernel(const unsigned short* __restrict__ A,
                      const unsigned short* __restrict__ Bw,
                      const float* __restrict__ bias,
                      float* __restrict__ Cout, int K) {
#if USE_ASYNC_LDS
#define NBUF 2
#else
#define NBUF 1
#endif
    __shared__ __align__(16) unsigned short sA[NBUF][128 * ASTR];
    __shared__ __align__(16) unsigned short sB[NBUF][64 * ASTR];

    const int tid  = threadIdx.x;
    const int lane = tid & 31;
    const int wave = tid >> 5;
    const int wm   = wave & 3;     // 0..3 -> M sub-block
    const int wn   = wave >> 2;    // 0..1 -> N sub-block
    const int row  = lane & 15;
    const int hi   = lane >> 4;

    const int m_blk = blockIdx.y * 128;
    const int n_blk = blockIdx.x * 64;

    // ---- copy-phase mapping: 16B chunks, row = tid>>3 (+32*s), q = tid&7 ----
    const int crow = tid >> 3;     // 0..31
    const int cq   = tid & 7;      // 0..7
    const unsigned short* gA = A  + (size_t)(m_blk + crow) * K + cq * 8;
    const unsigned short* gB = Bw + (size_t)(n_blk + crow) * K + cq * 8;
    const int ldsA = crow * ASTR + cq * 8;   // element offset in sA tile
    const int ldsB = crow * ASTR + cq * 8;

    v8f acc00 = {}, acc01 = {}, acc10 = {}, acc11 = {};

    const int nk = K >> 6;         // stages of KT=64

#if USE_ASYNC_LDS
    auto stage = [&](int buf, int k0) {
        // A: 128 rows x 128B = 1024 chunks -> 4 per thread
        #pragma unroll
        for (int s = 0; s < 4; ++s)
            GLD_ASYNC_B128(&sA[buf][ldsA + s * 32 * ASTR], gA + k0 + (size_t)s * 32 * K);
        // B: 64 rows x 128B = 512 chunks -> 2 per thread
        #pragma unroll
        for (int s = 0; s < 2; ++s)
            GLD_ASYNC_B128(&sB[buf][ldsB + s * 32 * ASTR], gB + k0 + (size_t)s * 32 * K);
    };
    stage(0, 0);
#endif

    for (int i = 0; i < nk; ++i) {
#if USE_ASYNC_LDS
        const int cur = i & 1;
        if (i + 1 < nk) {
            stage(cur ^ 1, (i + 1) << 6);
            WAIT_ASYNCCNT(6);      // next stage's 6 copies may stay in flight
        } else {
            WAIT_ASYNCCNT(0);
        }
        __syncthreads();           // all waves' copies for tile i complete
#else
        const int cur = 0;
        const int k0 = i << 6;
        uint4 ra[4], rb[2];
        #pragma unroll
        for (int s = 0; s < 4; ++s)
            ra[s] = *reinterpret_cast<const uint4*>(gA + k0 + (size_t)s * 32 * K);
        #pragma unroll
        for (int s = 0; s < 2; ++s)
            rb[s] = *reinterpret_cast<const uint4*>(gB + k0 + (size_t)s * 32 * K);
        __syncthreads();           // previous tile's readers done
        #pragma unroll
        for (int s = 0; s < 4; ++s)
            *reinterpret_cast<uint4*>(&sA[0][ldsA + s * 32 * ASTR]) = ra[s];
        #pragma unroll
        for (int s = 0; s < 2; ++s)
            *reinterpret_cast<uint4*>(&sB[0][ldsB + s * 32 * ASTR]) = rb[s];
        __syncthreads();
#endif

        const unsigned short* sAc = sA[cur];
        const unsigned short* sBc = sB[cur];
        #pragma unroll
        for (int j = 0; j < 2; ++j) {          // two K=32 sub-steps per stage
            v16bf a0, a1, b0, b1;
            // A fragment (16x32): lane row = M; elems 0..7 at K=j*32+hi*8,
            // elems 8..15 at K=j*32+16+hi*8   (CDNA5 ISA 7.12.2)
            {
                const int base = (wm * 32 + row) * ASTR + j * 32;
                uint4* p = reinterpret_cast<uint4*>(&a0);
                p[0] = *reinterpret_cast<const uint4*>(&sAc[base + hi * 8]);
                p[1] = *reinterpret_cast<const uint4*>(&sAc[base + 16 + hi * 8]);
            }
            {
                const int base = (wm * 32 + 16 + row) * ASTR + j * 32;
                uint4* p = reinterpret_cast<uint4*>(&a1);
                p[0] = *reinterpret_cast<const uint4*>(&sAc[base + hi * 8]);
                p[1] = *reinterpret_cast<const uint4*>(&sAc[base + 16 + hi * 8]);
            }
            // B fragment (32x16): lane row = N column; elems 0..7 at K=j*32+hi*8,
            // elems 8..15 at K=j*32+16+hi*8 (same as A)
            {
                const int base = (wn * 32 + row) * ASTR + j * 32;
                uint4* p = reinterpret_cast<uint4*>(&b0);
                p[0] = *reinterpret_cast<const uint4*>(&sBc[base + hi * 8]);
                p[1] = *reinterpret_cast<const uint4*>(&sBc[base + 16 + hi * 8]);
            }
            {
                const int base = (wn * 32 + 16 + row) * ASTR + j * 32;
                uint4* p = reinterpret_cast<uint4*>(&b1);
                p[0] = *reinterpret_cast<const uint4*>(&sBc[base + hi * 8]);
                p[1] = *reinterpret_cast<const uint4*>(&sBc[base + 16 + hi * 8])
            }

            acc00 = __builtin_amdgcn_wmma_f32_16x16x32_bf16(false, a0, false, b0, (short)0, acc00, false, false);
            acc01 = __builtin_amdgcn_wmma_f32_16x16x32_bf16(false, a0, false, b1, (short)0, acc01, false, false);
            acc10 = __builtin_amdgcn_wmma_f32_16x16x32_bf16(false, a1, false, b0, (short)0, acc10, false, false);
            acc11 = __builtin_amdgcn_wmma_f32_16x16x32_bf16(false, a1, false, b1, (short)0, acc11, false, false);
        }
#if USE_ASYNC_LDS
        __syncthreads();           // readers done before buffer is overwritten
#endif
    }

    // Epilogue: C layout (16x16 f32): lane col N = row, VGPR r -> M = r + hi*8
    const int n0 = n_blk + wn * 32;
    const int m0 = m_blk + wm * 32 + hi * 8;
    const float bn0 = bias[n0 + row];
    const float bn1 = bias[n0 + 16 + row];
    #pragma unroll
    for (int r = 0; r < 8; ++r) {
        Cout[(size_t)(m0 + r)      * NG + (n0 + row)]      = acc00[r] + bn0;
        Cout[(size_t)(m0 + r)      * NG + (n0 + 16 + row)] = acc01[r] + bn1;
        Cout[(size_t)(m0 + 16 + r) * NG + (n0 + row)]      = acc10[r] + bn0;
        Cout[(size_t)(m0 + 16 + r) * NG + (n0 + 16 + row)] = acc11[r] + bn1;
    }
}

// ---------------------------------------------------------------------------
// LSTM cell elementwise: gates order i,f,g,o. Writes bf16 h into next-GEMM
// operand buffer(s); optionally fp32 h (layer-2 -> d_out).
// ---------------------------------------------------------------------------
__global__ void lstm_cell_kernel(const float* __restrict__ gates,
                                 float* __restrict__ c,
                                 unsigned short* __restrict__ hdst1, int ld1, int off1,
                                 unsigned short* __restrict__ hdst2, int ld2, int off2,
                                 float* __restrict__ fout, int total) {
    int idx = blockIdx.x * blockDim.x + threadIdx.x;
    if (idx >= total) return;
    int bn = idx >> 9, n = idx & (RNN - 1);
    const float* g = gates + (size_t)bn * NG;
    float gi = sigmoidf(g[n]);
    float gf = sigmoidf(g[RNN + n]);
    float gg = tanhf(g[2 * RNN + n]);
    float go = sigmoidf(g[3 * RNN + n]);
    float cn = gf * c[idx] + gi * gg;
    float hn = go * tanhf(cn);
    c[idx] = cn;
    unsigned short hb = f2bf(hn);
    hdst1[(size_t)bn * ld1 + off1 + n] = hb;
    if (hdst2) hdst2[(size_t)bn * ld2 + off2 + n] = hb;
    if (fout)  fout[idx] = hn;
}

// ---------------------------------------------------------------------------
// Workspace layout (bytes, 256B-aligned; zero-init regions contiguous)
// ---------------------------------------------------------------------------
#define WC1_OFF    ((size_t)0)                                    // 3,407,872
#define WC2_OFF    ((size_t)3407872)                              // 4,194,304
#define B1_OFF     ((size_t)7602176)                              // 8,192
#define B2_OFF     ((size_t)7610368)                              // 8,192
#define GATES_OFF  ((size_t)7618560)                              // 104,857,600
#define Z1_OFF     ((size_t)112476160)                            // 21,299,200
#define Z2_OFF     ((size_t)133775360)                            // 26,214,400
#define C1_OFF     ((size_t)159989760)                            // 26,214,400
#define C2_OFF     ((size_t)186204160)                            // 26,214,400
#define ZERO_BYTES ((size_t)(212418560 - Z1_OFF))                 // 99,942,400

extern "C" void kernel_launch(void* const* d_in, const int* in_sizes, int n_in,
                              void* d_out, int out_size, void* d_ws, size_t ws_size,
                              hipStream_t stream) {
    const int*   sent = (const int*)  d_in[0];
    const float* w2v  = (const float*)d_in[1];
    const float* Wih1 = (const float*)d_in[2];
    const float* Whh1 = (const float*)d_in[3];
    const float* bih1 = (const float*)d_in[4];
    const float* bhh1 = (const float*)d_in[5];
    const float* Wih2 = (const float*)d_in[6];
    const float* Whh2 = (const float*)d_in[7];
    const float* bih2 = (const float*)d_in[8];
    const float* bhh2 = (const float*)d_in[9];
    float* out = (float*)d_out;

    char* ws = (char*)d_ws;
    unsigned short* Wc1   = (unsigned short*)(ws + WC1_OFF);
    unsigned short* Wc2   = (unsigned short*)(ws + WC2_OFF);
    float*          b1    = (float*)(ws + B1_OFF);
    float*          b2    = (float*)(ws + B2_OFF);
    float*          gates = (float*)(ws + GATES_OFF);
    unsigned short* Z1    = (unsigned short*)(ws + Z1_OFF);
    unsigned short* Z2    = (unsigned short*)(ws + Z2_OFF);
    float*          c1    = (float*)(ws + C1_OFF);
    float*          c2    = (float*)(ws + C2_OFF);

    const int TB = 256;

    {
        int tot = NG * K1;
        pack_w1_kernel<<<(tot + TB - 1) / TB, TB, 0, stream>>>(Wih1, Whh1, Wc1, tot);
    }
    {
        int tot = NG * K2;
        pack_w2_kernel<<<(tot + TB - 1) / TB, TB, 0, stream>>>(Wih2, Whh2, Wc2, tot);
    }
    bias_combine_kernel<<<(NG + TB - 1) / TB, TB, 0, stream>>>(bih1, bhh1, bih2, bhh2, b1, b2);
    {
        int tot = (int)(ZERO_BYTES / 16);
        zero16_kernel<<<(tot + TB - 1) / TB, TB, 0, stream>>>((uint4*)(ws + Z1_OFF), tot);
    }

    const dim3 gemm_grid(NG / 64, BN / 128);   // (32, 100)
    const int cell_tot  = BN * RNN;
    const int cell_blks = (cell_tot + TB - 1) / TB;
    const int emb_tot   = BN * EMB;
    const int emb_blks  = (emb_tot + TB - 1) / TB;

    for (int t = 0; t < TSTEPS; ++t) {
        embed_tanh_kernel<<<emb_blks, TB, 0, stream>>>(sent, w2v, Z1, t, emb_tot);

        // Layer 1: gates = [x_t | h1] @ [W_ih1 | W_hh1]^T + b1
        gemm_bf16_wmma_kernel<<<gemm_grid, TB, 0, stream>>>(Z1, Wc1, b1, gates, K1);
        lstm_cell_kernel<<<cell_blks, TB, 0, stream>>>(gates, c1,
                                                       Z1, K1, EMBP,
                                                       Z2, K2, 0,
                                                       (float*)nullptr, cell_tot);

        // Layer 2: gates = [h1 | h2] @ [W_ih2 | W_hh2]^T + b2
        gemm_bf16_wmma_kernel<<<gemm_grid, TB, 0, stream>>>(Z2, Wc2, b2, gates, K2);
        lstm_cell_kernel<<<cell_blks, TB, 0, stream>>>(gates, c2,
                                                       Z2, K2, RNN,
                                                       (unsigned short*)nullptr, 0, 0,
                                                       out, cell_tot);
    }
}
